// GRU_47364899340814
// MI455X (gfx1250) — compile-verified, matrix operand order checked by reading, not run on hardware
//
#include <hip/hip_runtime.h>
#include <math.h>

// GRU: B=128, T=512, D=64, H=512, two layers.
// Strategy:
//  1) cvt weights/activations to bf16 (matmul inputs only; all accumulation fp32)
//  2) xg = x @ Wx^T + bx  as a big parallel WMMA GEMM (time-parallel, done per layer)
//  3) persistent recurrent kernel: 256 blocks = 8 batch-row tiles x 32 hidden-col tiles,
//     each block keeps its 48KB Wh slice in LDS for all 512 steps (filled with
//     GLOBAL_LOAD_ASYNC_TO_LDS_B128 + s_wait_asynccnt); per-step sync is an
//     8-way-partitioned global atomic barrier (only blocks sharing batch rows interact),
//     h is double-buffered in bf16 for cross-block reads, fp32 h kept block-private.
//     Next step's xg rows are prefetched (global_prefetch_b8) under the WMMA chain.

typedef __attribute__((ext_vector_type(16))) __bf16 v16bf;
typedef __attribute__((ext_vector_type(8)))  __bf16 v8bf;
typedef __attribute__((ext_vector_type(8)))  float  v8f;

#define GB  128
#define GT  512
#define GD  64
#define GH  512
#define G3H 1536
#define BT  (GB * GT)
#define LDP 520   // padded LDS row stride (elems): 520*2B=1040B -> 4-bank rotation/row

// Load one 16x32 bf16 WMMA operand tile (A-style or B-style) from a row-major
// [rows x ldk] matrix. Per ISA 7.12.2 (16-bit 16x32 layout):
//   lane L (0..15) holds row L; lane half selects K chunk; v16bf elems 0..7 are
//   k0+half*8 .. +7 (contiguous), elems 8..15 are k0+16+half*8 .. +7.
static __device__ __forceinline__ v16bf load_tile(const __bf16* p, int row_base,
                                                  int k0, int ldk) {
  const int lane = threadIdx.x & 31;
  const __bf16* r = p + (size_t)(row_base + (lane & 15)) * ldk
                      + (k0 + ((lane >> 4) << 3));
  v8bf lo = *(const v8bf*)(r);
  v8bf hi = *(const v8bf*)(r + 16);
  return __builtin_shufflevector(lo, hi, 0, 1, 2, 3, 4, 5, 6, 7,
                                         8, 9, 10, 11, 12, 13, 14, 15);
}

static __device__ __forceinline__ v8f wmma_bf16(v16bf a, v16bf b, v8f c) {
  return __builtin_amdgcn_wmma_f32_16x16x32_bf16(false, a, false, b,
                                                 (short)0, c, false, false);
}

// Async bulk copy global -> LDS, 16B per lane per issue (ASYNCcnt-tracked DMA,
// no VGPR staging). dsoff = byte offset into the workgroup's LDS allocation.
static __device__ __forceinline__ void async_g2l_b128(unsigned dsoff,
                                                      const void* gsrc) {
  asm volatile("global_load_async_to_lds_b128 %0, %1, off"
               :: "v"(dsoff), "v"((unsigned long long)(uintptr_t)gsrc)
               : "memory");
}
static __device__ __forceinline__ void wait_asynccnt0() {
  asm volatile("s_wait_asynccnt 0x0" ::: "memory");
}

// ---------------------------------------------------------------- utilities
__global__ void cvt_bf16(const float* __restrict__ in, __bf16* __restrict__ out,
                         size_t n) {
  size_t i = (size_t)blockIdx.x * blockDim.x + threadIdx.x;
  size_t s = (size_t)gridDim.x * blockDim.x;
  for (; i < n; i += s) out[i] = (__bf16)in[i];
}

__global__ void zero_u32(unsigned* __restrict__ p, size_t n) {
  size_t i = (size_t)blockIdx.x * blockDim.x + threadIdx.x;
  size_t s = (size_t)gridDim.x * blockDim.x;
  for (; i < n; i += s) p[i] = 0u;
}

// ------------------------------------------------- time-parallel input GEMM
// out[m, g*H+c] = sum_k A[m,k] * W[g*H+c, k] + bx[g*H+c]
// A: [BT x K] bf16 row-major; W: [3H x K] bf16 row-major; out fp32 [BT x 3H].
// Block = 128 threads (4 waves); wave w computes N-tile blockIdx.y*4+w.
__global__ void __launch_bounds__(128) xg_gemm(const __bf16* __restrict__ A,
                                               const __bf16* __restrict__ W,
                                               const float* __restrict__ bx,
                                               float* __restrict__ out, int K) {
  const int lane = threadIdx.x & 31;
  const int wv   = threadIdx.x >> 5;
  const int mt   = blockIdx.x;
  const int nt   = blockIdx.y * 4 + wv;
  v8f acc = {};
  for (int k0 = 0; k0 < K; k0 += 32) {
    v16bf a = load_tile(A, mt * 16, k0, K);
    v16bf b = load_tile(W, nt * 16, k0, K);
    acc = wmma_bf16(a, b, acc);
  }
  const int n    = nt * 16 + (lane & 15);
  const float bi = bx[n];
  const int mrow = mt * 16 + ((lane >> 4) << 3);   // C/D: vgpr d -> M = d + 8*half
#pragma unroll
  for (int d = 0; d < 8; ++d)
    out[(size_t)(mrow + d) * G3H + n] = acc[d] + bi;
}

// ------------------------------------------------------ persistent recurrence
// grid = 256 blocks of 32 threads: rt = blockIdx.x>>5 (batch rows rt*16..+15),
// ct = blockIdx.x&31 (hidden cols ct*16..+15). Wh slice (3 gates x 16 cols x 512)
// lives in LDS for all T steps. h: fp32 copy is block-private (same rows+cols
// read & written); bf16 copy is double-buffered and shared within an rt group.
__global__ void __launch_bounds__(32) gru_rec(
    const float* __restrict__ xg, const __bf16* __restrict__ Wh,
    const float* __restrict__ bh, float* __restrict__ hf,
    __bf16* __restrict__ hbf, __bf16* __restrict__ ybf,
    float* __restrict__ yf, float* __restrict__ hlast,
    unsigned* __restrict__ bars) {
  __shared__ __bf16 ldsW[3 * 16 * LDP];            // ~49 KB
  const int lane = threadIdx.x;
  const int rt   = blockIdx.x >> 5;
  const int ct   = blockIdx.x & 31;

  // Async preload of this block's Wh slice: gate g, col c=ct*16+n -> row g*H+c.
  // Direct global->LDS DMA (no VGPR staging), drained with s_wait_asynccnt.
  for (int i = lane; i < (3 * 16 * GH) / 8; i += 32) {
    int e = i * 8;
    int r = e / GH;                                // 0..47 = g*16+n
    int k = e % GH;
    int g = r >> 4, n = r & 15;
    unsigned dsoff = (unsigned)(uintptr_t)(void*)&ldsW[(size_t)r * LDP + k];
    async_g2l_b128(dsoff, &Wh[((size_t)g * GH + ct * 16 + n) * GH + k]);
  }
  wait_asynccnt0();
  __syncthreads();

  const int col   = ct * 16 + (lane & 15);         // this lane's output column
  const float bhr = bh[col];
  const float bhz = bh[GH + col];
  const float bhn = bh[2 * GH + col];
  unsigned* bar   = bars + rt;
  const int NH    = GB * GH;

  for (int t = 0; t < GT; ++t) {
    const __bf16* cur = hbf + (size_t)(t & 1) * NH;
    __bf16* nxt       = hbf + (size_t)((t + 1) & 1) * NH;

    v8f ar = {}, az = {}, an = {};
    for (int kt = 0; kt < GH; kt += 32) {
      v16bf a  = load_tile(cur, rt * 16, kt, GH);          // h rows, reused 3x
      v16bf b0 = load_tile(ldsW + 0 * 16 * LDP, 0, kt, LDP);
      v16bf b1 = load_tile(ldsW + 1 * 16 * LDP, 0, kt, LDP);
      v16bf b2 = load_tile(ldsW + 2 * 16 * LDP, 0, kt, LDP);
      ar = wmma_bf16(a, b0, ar);
      az = wmma_bf16(a, b1, az);
      an = wmma_bf16(a, b2, an);
    }

#pragma unroll
    for (int d = 0; d < 8; ++d) {
      int m       = rt * 16 + d + ((lane >> 4) << 3);
      size_t xrow = ((size_t)m * GT + t) * G3H;            // xg row for (b=m, t)
      // Hide next step's xg latency under this step's tail + barrier.
      // (xg is read-only, so prefetching is race-free; h is NOT prefetched.)
      if (t + 1 < GT) {
        const float* nx = &xg[xrow + G3H];                 // (m, t+1) row
        __builtin_prefetch(nx + col, 0, 0);
        __builtin_prefetch(nx + GH + col, 0, 0);
        __builtin_prefetch(nx + 2 * GH + col, 0, 0);
      }
      float xr = xg[xrow + col];
      float xz = xg[xrow + GH + col];
      float xn = xg[xrow + 2 * GH + col];
      float r  = 1.f / (1.f + __expf(-(xr + ar[d] + bhr)));
      float z  = 1.f / (1.f + __expf(-(xz + az[d] + bhz)));
      float nn = tanhf(xn + r * (an[d] + bhn));
      size_t hidx = (size_t)m * GH + col;
      float ho = hf[hidx];
      float hv = z * ho + (1.f - z) * nn;
      hf[hidx]  = hv;
      nxt[hidx] = (__bf16)hv;
      size_t yidx = ((size_t)m * GT + t) * GH + col;
      if (ybf) ybf[yidx] = (__bf16)hv;
      if (yf)  yf[yidx]  = hv;
      if (hlast && t == GT - 1) hlast[hidx] = hv;
    }

    // Sync the 32 blocks sharing this rt group (monotonic counter barrier).
    if (t + 1 < GT) {
      __threadfence();
      if (lane == 0)
        __hip_atomic_fetch_add(bar, 1u, __ATOMIC_RELEASE, __HIP_MEMORY_SCOPE_AGENT);
      unsigned tgt = 32u * (unsigned)(t + 1);
      while (__hip_atomic_load(bar, __ATOMIC_ACQUIRE, __HIP_MEMORY_SCOPE_AGENT) < tgt)
        __builtin_amdgcn_s_sleep(1);
    }
  }
}

// --------------------------------------------------------------------- launch
extern "C" void kernel_launch(void* const* d_in, const int* in_sizes, int n_in,
                              void* d_out, int out_size, void* d_ws, size_t ws_size,
                              hipStream_t stream) {
  const float* x   = (const float*)d_in[0];
  const float* Wx0 = (const float*)d_in[1];
  const float* bx0 = (const float*)d_in[2];
  const float* Wh0 = (const float*)d_in[3];
  const float* bh0 = (const float*)d_in[4];
  const float* Wx1 = (const float*)d_in[5];
  const float* bx1 = (const float*)d_in[6];
  const float* Wh1 = (const float*)d_in[7];
  const float* bh1 = (const float*)d_in[8];
  float* y1    = (float*)d_out;
  float* hlast = y1 + (size_t)BT * GH;

  char* w = (char*)d_ws;
  auto alloc = [&](size_t bytes) {
    char* p = w;
    w += (bytes + 255) & ~(size_t)255;
    return p;
  };
  __bf16* xbf    = (__bf16*)alloc((size_t)BT * GD * 2);
  __bf16* wx0b   = (__bf16*)alloc((size_t)G3H * GD * 2);
  __bf16* wh0b   = (__bf16*)alloc((size_t)G3H * GH * 2);
  __bf16* wx1b   = (__bf16*)alloc((size_t)G3H * GH * 2);
  __bf16* wh1b   = (__bf16*)alloc((size_t)G3H * GH * 2);
  __bf16* y0bf   = (__bf16*)alloc((size_t)BT * GH * 2);
  float*  xg     = (float*) alloc((size_t)BT * G3H * 4);   // reused by both layers
  float*  hf     = (float*) alloc((size_t)GB * GH * 4);    // |
  __bf16* hbf    = (__bf16*)alloc((size_t)2 * GB * GH * 2);//  > contiguous zero region
  unsigned* bars = (unsigned*)alloc(256);                  // |

  // bf16 conversions
  cvt_bf16<<<512, 256, 0, stream>>>(x,   xbf,  (size_t)BT * GD);
  cvt_bf16<<<128, 256, 0, stream>>>(Wx0, wx0b, (size_t)G3H * GD);
  cvt_bf16<<<512, 256, 0, stream>>>(Wh0, wh0b, (size_t)G3H * GH);
  cvt_bf16<<<512, 256, 0, stream>>>(Wx1, wx1b, (size_t)G3H * GH);
  cvt_bf16<<<512, 256, 0, stream>>>(Wh1, wh1b, (size_t)G3H * GH);

  // zero h (fp32 + both bf16 buffers) + all barrier counters (contiguous)
  size_t zwords = ((size_t)GB * GH * 4 + (size_t)2 * GB * GH * 2 + 256) / 4;
  zero_u32<<<256, 256, 0, stream>>>((unsigned*)hf, zwords);

  // ---- layer 0
  xg_gemm<<<dim3(BT / 16, G3H / 64), 128, 0, stream>>>(xbf, wx0b, bx0, xg, GD);
  gru_rec<<<256, 32, 0, stream>>>(xg, wh0b, bh0, hf, hbf, y0bf, nullptr, nullptr,
                                  bars);

  // ---- reset h for layer 1 (leave bars[8..15] at zero from the first clear)
  zero_u32<<<256, 256, 0, stream>>>((unsigned*)hf,
                                    ((size_t)GB * GH * 4 + (size_t)2 * GB * GH * 2) / 4);
  xg_gemm<<<dim3(BT / 16, G3H / 64), 128, 0, stream>>>(y0bf, wx1b, bx1, xg, GH);
  gru_rec<<<256, 32, 0, stream>>>(xg, wh1b, bh1, hf, hbf, nullptr, y1, hlast,
                                  bars + 8);
}
